// LinearTemporalCrossAttention_53446573031455
// MI455X (gfx1250) — matrix operand order repaired and checked by reading, not compiled, not run on hardware
//
#include <hip/hip_runtime.h>
#include <hip/hip_bf16.h>

typedef __bf16 bf16;
typedef __attribute__((ext_vector_type(16))) __bf16 v16bf;
typedef __attribute__((ext_vector_type(8)))  __bf16 v8bf;
typedef __attribute__((ext_vector_type(8)))  float   v8f;

#define B_  64
#define T_  1024
#define N_  77
#define D_  512
#define H_  8
#define DT_ 256
#define DE_ 2048
#define DH_ 64
#define ROWS_BT (B_*T_)      // 65536
#define ROWS_BN (B_*N_)      // 4928

static __device__ __forceinline__ v16bf frag_join(v8bf lo, v8bf hi){
  v16bf r;
#pragma unroll
  for(int i=0;i<8;++i){ r[i]=lo[i]; r[i+8]=hi[i]; }
  return r;
}

// A-fragment (16x32, MxK, bf16) from row-major [rows][pitch] buffer.
// Lane l (0..15) holds row mbase+l; half=lane>>4 selects K-chunk:
//   element i -> K = kbase + half*16 + i (16 contiguous)
static __device__ __forceinline__ v16bf load_a(const bf16* base, int pitch,
                                               int mbase, int kbase, int l, int half){
  const bf16* p = base + (size_t)(mbase + l)*pitch + kbase + half*16;
  v8bf lo = *(const v8bf*)(p);
  v8bf hi = *(const v8bf*)(p + 8);
  return frag_join(lo, hi);
}

// B-fragment (32x16, KxN) where memory is row-major W[N][K] (B = W^T):
// lane l -> column n; element i -> K = kbase + half*16 + i (16 contiguous bf16).
static __device__ __forceinline__ v16bf load_bT(const bf16* W, int ldk, int n,
                                                int kbase, int half){
  return *(const v16bf*)(W + (size_t)n*ldk + kbase + half*16);
}

static __device__ __forceinline__ v8f wmma_bf16(v16bf a, v16bf b, v8f c){
  return __builtin_amdgcn_wmma_f32_16x16x32_bf16(false, a, false, b, (short)0, c,
                                                 false, false);
}

static __device__ __forceinline__ float silu_f(float x){
  return x / (1.f + __expf(-x));
}

// ---------------- prep kernels -------------------------------------------
__global__ __launch_bounds__(256) void cvt_bf16_kernel(const float* __restrict__ s,
                                                       bf16* __restrict__ d, int n){
  int i = blockIdx.x*256 + threadIdx.x;
  if(i < n) d[i] = (bf16)s[i];
}

__global__ __launch_bounds__(256) void silu_bf16_kernel(const float* __restrict__ s,
                                                        bf16* __restrict__ d, int n){
  int i = blockIdx.x*256 + threadIdx.x;
  if(i < n) d[i] = (bf16)silu_f(s[i]);
}

// ---------------- K/V projection: LN(xf) -> kpre/vpre --------------------
// grid: ceil(4928/32) blocks x 256 threads (8 waves). Mtile=32, Ntile=512.
__global__ __launch_bounds__(256) void kv_proj_kernel(
    const float* __restrict__ xf, const float* __restrict__ ln_t_w,
    const float* __restrict__ ln_t_b,
    const bf16* __restrict__ Wk_bf, const float* __restrict__ bk,
    const bf16* __restrict__ Wv_bf, const float* __restrict__ bv,
    float* __restrict__ kpre, float* __restrict__ vpre)
{
  __shared__ bf16 tbuf[32*272];          // 32 rows x (256+16) bf16
  const int tid = threadIdx.x, lane = tid & 31, w = tid >> 5;
  const int l = lane & 15, half = lane >> 4;
  const int rowbase = blockIdx.x * 32;

  // stage0: LayerNorm over DT=256, write bf16 to LDS (zeros for OOB rows)
  for(int rr = 0; rr < 4; ++rr){
    int r = w*4 + rr, grow = rowbase + r;
    float vals[8]; float s = 0.f, ss = 0.f;
    if(grow < ROWS_BN){
      const float* xr = xf + (size_t)grow * DT_;
#pragma unroll
      for(int i=0;i<8;++i){ float v = xr[i*32 + lane]; vals[i]=v; s+=v; ss+=v*v; }
    } else {
#pragma unroll
      for(int i=0;i<8;++i) vals[i]=0.f;
    }
#pragma unroll
    for(int off=16; off>=1; off>>=1){ s += __shfl_xor(s,off,32); ss += __shfl_xor(ss,off,32); }
    float mu = s * (1.f/DT_);
    float rsig = rsqrtf(ss * (1.f/DT_) - mu*mu + 1e-5f);
#pragma unroll
    for(int i=0;i<8;++i){
      int n = i*32 + lane;
      float xn = (grow < ROWS_BN) ? (vals[i]-mu)*rsig*ln_t_w[n] + ln_t_b[n] : 0.f;
      tbuf[r*272 + n] = (bf16)xn;
    }
  }
  __syncthreads();

  const int rt = w >> 2, cq = w & 3;     // row-tile 0..1, col-quarter 0..3
  // --- kpre = tn @ Wk^T + bk ---
  {
    v8f acc[8];
#pragma unroll
    for(int t=0;t<8;++t) acc[t] = (v8f){0,0,0,0,0,0,0,0};
    for(int kk=0; kk<8; ++kk){
      v16bf a = load_a(tbuf, 272, rt*16, kk*32, l, half);
#pragma unroll
      for(int t=0;t<8;++t){
        int n = cq*128 + t*16 + l;
        acc[t] = wmma_bf16(a, load_bT(Wk_bf, DT_, n, kk*32, half), acc[t]);
      }
    }
#pragma unroll
    for(int t=0;t<8;++t){
      int n = cq*128 + t*16 + l; float bias = bk[n];
#pragma unroll
      for(int r=0;r<8;++r){
        int grow = rowbase + rt*16 + r + half*8;
        if(grow < ROWS_BN) kpre[(size_t)grow*D_ + n] = acc[t][r] + bias;
      }
    }
  }
  // --- vpre = tn @ Wv^T + bv ---
  {
    v8f acc[8];
#pragma unroll
    for(int t=0;t<8;++t) acc[t] = (v8f){0,0,0,0,0,0,0,0};
    for(int kk=0; kk<8; ++kk){
      v16bf a = load_a(tbuf, 272, rt*16, kk*32, l, half);
#pragma unroll
      for(int t=0;t<8;++t){
        int n = cq*128 + t*16 + l;
        acc[t] = wmma_bf16(a, load_bT(Wv_bf, DT_, n, kk*32, half), acc[t]);
      }
    }
#pragma unroll
    for(int t=0;t<8;++t){
      int n = cq*128 + t*16 + l; float bias = bv[n];
#pragma unroll
      for(int r=0;r<8;++r){
        int grow = rowbase + rt*16 + r + half*8;
        if(grow < ROWS_BN) vpre[(size_t)grow*D_ + n] = acc[t][r] + bias;
      }
    }
  }
}

// ---------------- attnT[b,h] = (softmax_n(k)^T v)^T ----------------------
// grid: B*H blocks x 128 threads. Stores attnT[dcol][d] bf16 so the main
// kernel's B-fragments are contiguous in d. Softmax 1/sum folded into store.
__global__ __launch_bounds__(128) void attn_kernel(
    const float* __restrict__ kpre, const float* __restrict__ vpre,
    bf16* __restrict__ attnT)
{
  __shared__ bf16 kbT[64*104];   // [d][n] : exp(k - max), n padded 77->96
  __shared__ bf16 vbT[64*104];   // [d][n] : v
  __shared__ float sden[64];     // 1/sum per d
  const int bh = blockIdx.x, b = bh >> 3, h = bh & 7;
  const int tid = threadIdx.x;
  const float* kp = kpre + (size_t)b*N_*D_ + h*DH_;
  const float* vp = vpre + (size_t)b*N_*D_ + h*DH_;

  if(tid < 64){
    int d = tid;
    float m = -1e30f;
    for(int n=0;n<N_;++n) m = fmaxf(m, kp[(size_t)n*D_ + d]);
    float s = 0.f;
    for(int n=0;n<N_;++n){
      float e = __expf(kp[(size_t)n*D_ + d] - m);
      s += e;
      kbT[d*104 + n] = (bf16)e;
      vbT[d*104 + n] = (bf16)vp[(size_t)n*D_ + d];
    }
    for(int n=N_;n<96;++n){ kbT[d*104+n] = (bf16)0.f; vbT[d*104+n] = (bf16)0.f; }
    sden[d] = 1.f / s;
  }
  __syncthreads();

  const int lane = tid & 31, w = tid >> 5;
  const int l = lane & 15, half = lane >> 4;
  const int mt = w;                          // m-tile (dcol) 0..3
  v8f acc[4];
#pragma unroll
  for(int t=0;t<4;++t) acc[t] = (v8f){0,0,0,0,0,0,0,0};
  for(int kk=0; kk<3; ++kk){                 // K = 96 (padded N)
    // A[m=dcol][k=n] = v[n][dcol] = vbT[dcol][n]
    v16bf a = load_a(vbT, 104, mt*16, kk*32, l, half);
#pragma unroll
    for(int t=0;t<4;++t){
      int d = t*16 + l;                      // B[k=n][col=d] = kbT[d][n]
      const bf16* p = kbT + d*104 + kk*32 + half*16;
      v8bf lo = *(const v8bf*)(p);
      v8bf hi = *(const v8bf*)(p + 8);
      acc[t] = wmma_bf16(a, frag_join(lo,hi), acc[t]);
    }
  }
  bf16* out = attnT + (size_t)bh * DH_ * DH_;
#pragma unroll
  for(int t=0;t<4;++t){
    int d = t*16 + l; float inv = sden[d];
#pragma unroll
    for(int r=0;r<8;++r){
      int dcol = mt*16 + r + half*8;
      out[dcol*DH_ + d] = (bf16)(acc[t][r] * inv);
    }
  }
}

// ---------------- emb_out = silu(emb) @ emb_w^T + emb_b ------------------
// grid: 16 blocks x 256 threads; block covers all 64 rows x 64 cols.
__global__ __launch_bounds__(256) void emb_mlp_kernel(
    const bf16* __restrict__ semb, const bf16* __restrict__ embw_bf,
    const float* __restrict__ emb_b, float* __restrict__ emb_out)
{
  const int tid = threadIdx.x, lane = tid & 31, w = tid >> 5;
  const int l = lane & 15, half = lane >> 4;
  const int colbase = blockIdx.x * 64;
  const int rt = w >> 1, cp = w & 1;
  v8f acc[2] = { (v8f){0,0,0,0,0,0,0,0}, (v8f){0,0,0,0,0,0,0,0} };
  for(int kk=0; kk<64; ++kk){
    v16bf a = load_a(semb, DE_, rt*16, kk*32, l, half);
#pragma unroll
    for(int t=0;t<2;++t){
      int n = colbase + (cp*2 + t)*16 + l;
      acc[t] = wmma_bf16(a, load_bT(embw_bf, DE_, n, kk*32, half), acc[t]);
    }
  }
#pragma unroll
  for(int t=0;t<2;++t){
    int n = colbase + (cp*2 + t)*16 + l; float bias = emb_b[n];
#pragma unroll
    for(int r=0;r<8;++r){
      int m = rt*16 + r + half*8;
      emb_out[m*1024 + n] = acc[t][r] + bias;
    }
  }
}

// ---------------- fused main: LN -> Q -> softmax -> y -> stylize -> out --
// grid: 2048 blocks x 256 threads (8 waves). Mtile=32 rows, Ntile=512 cols.
__global__ __launch_bounds__(256) void fused_main_kernel(
    const float* __restrict__ x, const float* __restrict__ ln_x_w,
    const float* __restrict__ ln_x_b,
    const bf16* __restrict__ Wq_bf, const float* __restrict__ bq,
    const bf16* __restrict__ attnT, const float* __restrict__ emb_out,
    const float* __restrict__ ln_s_w, const float* __restrict__ ln_s_b,
    const bf16* __restrict__ outw_bf, const float* __restrict__ out_b,
    const float* __restrict__ gate, float* __restrict__ out)
{
  __shared__ bf16 buf[32*528];             // reused: xn -> q -> silu(h)
  __shared__ float ssum[32], ssq[32];
  const int tid = threadIdx.x, lane = tid & 31, w = tid >> 5;
  const int l = lane & 15, half = lane >> 4;
  const int rowbase = blockIdx.x * 32;
  const int bb = rowbase >> 10;            // batch index (T=1024, 32 | 1024)

  // ---- stage0: xn = LN(x) -> LDS bf16 ----
  for(int rr=0; rr<4; ++rr){
    int r = w*4 + rr;
    const float* xr = x + (size_t)(rowbase + r) * D_;
    float vals[16]; float s = 0.f, ss = 0.f;
#pragma unroll
    for(int i=0;i<16;++i){ float v = xr[i*32 + lane]; vals[i]=v; s+=v; ss+=v*v; }
#pragma unroll
    for(int off=16; off>=1; off>>=1){ s += __shfl_xor(s,off,32); ss += __shfl_xor(ss,off,32); }
    float mu = s * (1.f/D_);
    float rsig = rsqrtf(ss * (1.f/D_) - mu*mu + 1e-5f);
#pragma unroll
    for(int i=0;i<16;++i){
      int n = i*32 + lane;
      buf[r*528 + n] = (bf16)((vals[i]-mu)*rsig*ln_x_w[n] + ln_x_b[n]);
    }
  }
  if(tid < 32){ ssum[tid] = 0.f; ssq[tid] = 0.f; }
  __syncthreads();

  const int rt = w >> 2, cq = w & 3;       // row-tile 0..1, col-quarter 0..3
  // ---- GEMM1: qpre = xn @ Wq^T ----
  v8f acc[8];
#pragma unroll
  for(int t=0;t<8;++t) acc[t] = (v8f){0,0,0,0,0,0,0,0};
  for(int kk=0; kk<16; ++kk){
    v16bf a = load_a(buf, 528, rt*16, kk*32, l, half);
    if(kk < 15)
      __builtin_prefetch(Wq_bf + (size_t)(cq*128 + l)*D_ + (kk+1)*32, 0, 3);
#pragma unroll
    for(int t=0;t<8;++t){
      int n = cq*128 + t*16 + l;
      acc[t] = wmma_bf16(a, load_bT(Wq_bf, D_, n, kk*32, half), acc[t]);
    }
  }
  // ---- bias + per-head softmax over Dh=64 (4 tiles x 16 lanes/half) ----
#pragma unroll
  for(int t=0;t<8;++t){
    int n = cq*128 + t*16 + l; float bias = bq[n];
#pragma unroll
    for(int r=0;r<8;++r) acc[t][r] += bias;
  }
#pragma unroll
  for(int g=0; g<2; ++g){
#pragma unroll
    for(int r=0;r<8;++r){
      float m = -1e30f;
#pragma unroll
      for(int t=0;t<4;++t) m = fmaxf(m, acc[g*4+t][r]);
#pragma unroll
      for(int off=8; off>=1; off>>=1) m = fmaxf(m, __shfl_xor(m, off, 16));
      float s = 0.f;
#pragma unroll
      for(int t=0;t<4;++t){ float e = __expf(acc[g*4+t][r]-m); acc[g*4+t][r]=e; s+=e; }
#pragma unroll
      for(int off=8; off>=1; off>>=1) s += __shfl_xor(s, off, 16);
      float inv = 1.f / s;
#pragma unroll
      for(int t=0;t<4;++t) acc[g*4+t][r] *= inv;
    }
  }
  __syncthreads();                          // xn no longer needed
  // ---- q -> LDS (bf16, overwrite buf) ----
#pragma unroll
  for(int t=0;t<8;++t){
    int n = cq*128 + t*16 + l;
#pragma unroll
    for(int r=0;r<8;++r){
      int m = rt*16 + r + half*8;
      buf[m*528 + n] = (bf16)acc[t][r];
    }
  }
  __syncthreads();
  // ---- GEMM2: y = q @ attn (per head, B from attnT, k-contiguous) ----
  v8f yacc[8];
#pragma unroll
  for(int t=0;t<8;++t) yacc[t] = (v8f){0,0,0,0,0,0,0,0};
#pragma unroll
  for(int g=0; g<2; ++g){
    int hh = cq*2 + g;
    const bf16* at = attnT + ((size_t)bb*H_ + hh) * DH_ * DH_;
#pragma unroll
    for(int kk=0; kk<2; ++kk){
      v16bf a = load_a(buf, 528, rt*16, hh*64 + kk*32, l, half);
#pragma unroll
      for(int t=0;t<4;++t){
        int dcol = t*16 + l;
        v16bf bfr = *(const v16bf*)(at + (size_t)dcol*DH_ + kk*32 + half*16);
        yacc[g*4+t] = wmma_bf16(a, bfr, yacc[g*4+t]);
      }
    }
  }
  // ---- LN stats of y (row sums via shfl + LDS float atomics) ----
#pragma unroll
  for(int r=0;r<8;++r){
    float s = 0.f, ss = 0.f;
#pragma unroll
    for(int t=0;t<8;++t){ float v = yacc[t][r]; s += v; ss += v*v; }
#pragma unroll
    for(int off=8; off>=1; off>>=1){ s += __shfl_xor(s,off,16); ss += __shfl_xor(ss,off,16); }
    if(l == 0){
      int m = rt*16 + r + half*8;
      atomicAdd(&ssum[m], s);
      atomicAdd(&ssq[m], ss);
    }
  }
  __syncthreads();
  // ---- stylization + silu -> LDS bf16 (overwrite buf) ----
#pragma unroll
  for(int r=0;r<8;++r){
    int m = rt*16 + r + half*8;
    float mu = ssum[m] * (1.f/D_);
    float rsig = rsqrtf(ssq[m] * (1.f/D_) - mu*mu + 1e-5f);
#pragma unroll
    for(int t=0;t<8;++t){
      int n = cq*128 + t*16 + l;
      float sc = 1.f + emb_out[bb*1024 + n];
      float sh = emb_out[bb*1024 + 512 + n];
      float hv = ((yacc[t][r]-mu)*rsig*ln_s_w[n] + ln_s_b[n]) * sc + sh;
      buf[m*528 + n] = (bf16)silu_f(hv);
    }
  }
  __syncthreads();
  // ---- GEMM3: h @ out_w^T ----
  v8f oacc[8];
#pragma unroll
  for(int t=0;t<8;++t) oacc[t] = (v8f){0,0,0,0,0,0,0,0};
  for(int kk=0; kk<16; ++kk){
    v16bf a = load_a(buf, 528, rt*16, kk*32, l, half);
    if(kk < 15)
      __builtin_prefetch(outw_bf + (size_t)(cq*128 + l)*D_ + (kk+1)*32, 0, 3);
#pragma unroll
    for(int t=0;t<8;++t){
      int n = cq*128 + t*16 + l;
      oacc[t] = wmma_bf16(a, load_bT(outw_bf, D_, n, kk*32, half), oacc[t]);
    }
  }
  // ---- epilogue: out = x + sigmoid(gate)*(h + out_b) ----
  float sg = 1.f / (1.f + __expf(-gate[0]));
#pragma unroll
  for(int t=0;t<8;++t){
    int n = cq*128 + t*16 + l; float bias = out_b[n];
#pragma unroll
    for(int r=0;r<8;++r){
      size_t grow = (size_t)rowbase + rt*16 + r + half*8;
      out[grow*D_ + n] = x[grow*D_ + n] + sg * (oacc[t][r] + bias);
    }
  }
}

// ---------------- host side ----------------------------------------------
extern "C" void kernel_launch(void* const* d_in, const int* in_sizes, int n_in,
                              void* d_out, int out_size, void* d_ws, size_t ws_size,
                              hipStream_t stream) {
  const float* x      = (const float*)d_in[0];
  const float* xf     = (const float*)d_in[1];
  const float* emb    = (const float*)d_in[2];
  const float* ln_x_w = (const float*)d_in[3];
  const float* ln_x_b = (const float*)d_in[4];
  const float* ln_t_w = (const float*)d_in[5];
  const float* ln_t_b = (const float*)d_in[6];
  const float* Wq     = (const float*)d_in[7];
  const float* bq     = (const float*)d_in[8];
  const float* Wk     = (const float*)d_in[9];
  const float* bk     = (const float*)d_in[10];
  const float* Wv     = (const float*)d_in[11];
  const float* bv     = (const float*)d_in[12];
  const float* gate   = (const float*)d_in[13];
  const float* emb_w  = (const float*)d_in[14];
  const float* emb_b  = (const float*)d_in[15];
  const float* ln_s_w = (const float*)d_in[16];
  const float* ln_s_b = (const float*)d_in[17];
  const float* out_w  = (const float*)d_in[18];
  const float* out_b  = (const float*)d_in[19];
  float* out = (float*)d_out;

  char* ws = (char*)d_ws;
  size_t off = 0;
  auto alloc = [&](size_t bytes){ void* p = ws + off; off += (bytes + 255) & ~size_t(255); return p; };
  bf16*  Wq_bf   = (bf16*) alloc((size_t)D_*D_*2);
  bf16*  Wk_bf   = (bf16*) alloc((size_t)D_*DT_*2);
  bf16*  Wv_bf   = (bf16*) alloc((size_t)D_*DT_*2);
  bf16*  outw_bf = (bf16*) alloc((size_t)D_*D_*2);
  bf16*  embw_bf = (bf16*) alloc((size_t)2*D_*DE_*2);
  bf16*  semb_bf = (bf16*) alloc((size_t)B_*DE_*2);
  float* kpre    = (float*)alloc((size_t)ROWS_BN*D_*4);
  float* vpre    = (float*)alloc((size_t)ROWS_BN*D_*4);
  bf16*  attnT   = (bf16*) alloc((size_t)B_*H_*DH_*DH_*2);
  float* emb_out = (float*)alloc((size_t)B_*2*D_*4);
  (void)ws_size; (void)n_in; (void)in_sizes; (void)out_size;

  // precision conversion (weights -> bf16), silu(emb) -> bf16
  cvt_bf16_kernel<<<(D_*D_+255)/256, 256, 0, stream>>>(Wq, Wq_bf, D_*D_);
  cvt_bf16_kernel<<<(D_*DT_+255)/256, 256, 0, stream>>>(Wk, Wk_bf, D_*DT_);
  cvt_bf16_kernel<<<(D_*DT_+255)/256, 256, 0, stream>>>(Wv, Wv_bf, D_*DT_);
  cvt_bf16_kernel<<<(D_*D_+255)/256, 256, 0, stream>>>(out_w, outw_bf, D_*D_);
  cvt_bf16_kernel<<<(2*D_*DE_+255)/256, 256, 0, stream>>>(emb_w, embw_bf, 2*D_*DE_);
  silu_bf16_kernel<<<(B_*DE_+255)/256, 256, 0, stream>>>(emb, semb_bf, B_*DE_);

  // temporal path
  kv_proj_kernel<<<(ROWS_BN + 31)/32, 256, 0, stream>>>(
      xf, ln_t_w, ln_t_b, Wk_bf, bk, Wv_bf, bv, kpre, vpre);
  attn_kernel<<<B_*H_, 128, 0, stream>>>(kpre, vpre, attnT);

  // conditioning path
  emb_mlp_kernel<<<(2*D_)/64, 256, 0, stream>>>(semb_bf, embw_bf, emb_b, emb_out);

  // fused main path
  fused_main_kernel<<<ROWS_BT/32, 256, 0, stream>>>(
      x, ln_x_w, ln_x_b, Wq_bf, bq, attnT, emb_out,
      ln_s_w, ln_s_b, outw_bf, out_b, gate, out);
}
